// FP8Linear_38843684225317
// MI455X (gfx1250) — 8-bit matrix operand order repaired and checked by reading, compiled, not run on hardware
//
#include <hip/hip_runtime.h>
#include <hip/hip_bf16.h>


typedef __attribute__((ext_vector_type(16))) int   v16i;
typedef __attribute__((ext_vector_type(8)))  float v8f;

#define BM 128
#define BN 128
#define BK 128
#define FP8_MAX 448.0f

#if defined(__has_builtin) && __has_builtin(__builtin_amdgcn_global_load_async_to_lds_b128)
#define HAVE_ASYNC_LDS 1
typedef int v4i_vs __attribute__((vector_size(16)));
typedef v4i_vs __attribute__((address_space(1)))* as1_v4i;
typedef v4i_vs __attribute__((address_space(3)))* as3_v4i;
#define ASYNC_COPY_B128(gptr, lptr)                                              \
    __builtin_amdgcn_global_load_async_to_lds_b128(                              \
        (as1_v4i)(unsigned long long)(gptr),                                     \
        (as3_v4i)(unsigned int)(unsigned long long)(lptr), 0, 0)
#if __has_builtin(__builtin_amdgcn_s_wait_asynccnt)
#define WAIT_ASYNC() __builtin_amdgcn_s_wait_asynccnt(0)
#else
#define WAIT_ASYNC() asm volatile("s_wait_asynccnt 0" ::: "memory")
#endif
#else
#define HAVE_ASYNC_LDS 0
#define ASYNC_COPY_B128(gptr, lptr) (*(int4*)(lptr) = *(const int4*)(gptr))
#define WAIT_ASYNC() do {} while (0)
#endif

__global__ void zero_scales_kernel(float* amax) {
    if (threadIdx.x < 2) amax[threadIdx.x] = 0.0f;
}

__global__ void amax_kernel(const float* __restrict__ in, long long n4,
                            float* __restrict__ out_amax) {
    __shared__ float red[256];
    const float4* in4 = (const float4*)in;
    float m = 0.0f;
    long long stride = (long long)gridDim.x * blockDim.x;
    for (long long i = (long long)blockIdx.x * blockDim.x + threadIdx.x; i < n4; i += stride) {
        float4 v = in4[i];
        m = fmaxf(m, fmaxf(fmaxf(fabsf(v.x), fabsf(v.y)), fmaxf(fabsf(v.z), fabsf(v.w))));
    }
    red[threadIdx.x] = m;
    __syncthreads();
    for (int s = 128; s > 0; s >>= 1) {
        if ((int)threadIdx.x < s) red[threadIdx.x] = fmaxf(red[threadIdx.x], red[threadIdx.x + s]);
        __syncthreads();
    }
    if (threadIdx.x == 0) {
        atomicMax((unsigned int*)out_amax, __float_as_uint(red[0]));
    }
}

__device__ inline unsigned char f32_to_e4m3_sw(float x) {
    unsigned int u = __float_as_uint(x);
    unsigned char sign = (unsigned char)((u >> 31) << 7);
    float a = fminf(fabsf(x), FP8_MAX);
    unsigned int bits = __float_as_uint(a);
    int exp = (int)((bits >> 23) & 0xFF) - 127;
    unsigned int mant = bits & 0x7FFFFF;
    unsigned char out;
    if (exp < -10) {
        out = 0;
    } else if (exp < -6) {
        int shift = -6 - exp;
        unsigned int m24 = mant | 0x800000u;
        int drop = 20 + shift;
        unsigned int keep = m24 >> drop;
        unsigned int rem  = m24 & ((1u << drop) - 1u);
        unsigned int half = 1u << (drop - 1);
        if (rem > half || (rem == half && (keep & 1u))) keep++;
        out = (unsigned char)keep;
    } else {
        unsigned int keep = mant >> 20;
        unsigned int rem  = mant & 0xFFFFFu;
        if (rem > 0x80000u || (rem == 0x80000u && (keep & 1u))) keep++;
        int e = exp + 7;
        if (keep > 7) { keep = 0; e++; }
        if (e > 15 || (e == 15 && keep > 6)) { e = 15; keep = 6; }
        out = (unsigned char)((e << 3) | keep);
    }
    return out | sign;
}

__global__ void quant_kernel(const float* __restrict__ in, unsigned char* __restrict__ out,
                             long long n4, const float* __restrict__ amax_p) {
    const float amax = fmaxf(*amax_p, 1e-12f);
    const float r = FP8_MAX / amax;
    const float4* in4 = (const float4*)in;
    unsigned int* out4 = (unsigned int*)out;
    long long stride = (long long)gridDim.x * blockDim.x;
    for (long long i = (long long)blockIdx.x * blockDim.x + threadIdx.x; i < n4; i += stride) {
        float4 v = in4[i];
        float a = fminf(fmaxf(v.x * r, -FP8_MAX), FP8_MAX);
        float b = fminf(fmaxf(v.y * r, -FP8_MAX), FP8_MAX);
        float c = fminf(fmaxf(v.z * r, -FP8_MAX), FP8_MAX);
        float d = fminf(fmaxf(v.w * r, -FP8_MAX), FP8_MAX);
        unsigned int packed;
#if __has_builtin(__builtin_amdgcn_cvt_pk_fp8_f32)
        packed = (unsigned int)__builtin_amdgcn_cvt_pk_fp8_f32(a, b, 0, false);
        packed = (unsigned int)__builtin_amdgcn_cvt_pk_fp8_f32(c, d, (int)packed, true);
#else
        packed = (unsigned int)f32_to_e4m3_sw(a)
               | ((unsigned int)f32_to_e4m3_sw(b) << 8)
               | ((unsigned int)f32_to_e4m3_sw(c) << 16)
               | ((unsigned int)f32_to_e4m3_sw(d) << 24);
#endif
        out4[i] = packed;
    }
}

union FragU {
    v16i v;
    int  d[16];
};

__global__ __launch_bounds__(256) void gemm_fp8_kernel(
    const unsigned char* __restrict__ Xq,
    const unsigned char* __restrict__ Wq,
    const float* __restrict__ bias,
    const float* __restrict__ amax,
    float* __restrict__ out,
    int M, int N, int K)
{
    __shared__ __align__(16) unsigned char Asmem[2][BM * BK];
    __shared__ __align__(16) unsigned char Bsmem[2][BN * BK];

    const int tid  = threadIdx.x;
    const int lane = tid & 31;
    const int wave = tid >> 5;
    const int wm   = wave >> 1;
    const int wn   = wave & 1;

    const int mrow = lane & 15;
    const int khi  = lane >> 4;

    const long long block_m = (long long)blockIdx.y * BM;
    const long long block_n = (long long)blockIdx.x * BN;

    const int c_row[4] = { (tid + 0)   >> 3, (tid + 256) >> 3,
                           (tid + 512) >> 3, (tid + 768) >> 3 };
    const int c_col    = (tid & 7) * 16;

    const v8f vzero = {0.f, 0.f, 0.f, 0.f, 0.f, 0.f, 0.f, 0.f};
    v8f acc[2][4];
#pragma unroll
    for (int i = 0; i < 2; ++i)
#pragma unroll
        for (int j = 0; j < 4; ++j) acc[i][j] = vzero;

#pragma unroll
    for (int i = 0; i < 4; ++i) {
        int row = c_row[i];
        ASYNC_COPY_B128(&Xq[(block_m + row) * (long long)K + c_col],
                        &Asmem[0][row * BK + c_col]);
        ASYNC_COPY_B128(&Wq[(block_n + row) * (long long)K + c_col],
                        &Bsmem[0][row * BK + c_col]);
    }
    WAIT_ASYNC();
    __syncthreads();

    int cur = 0;
    for (int k0 = 0; k0 < K; k0 += BK) {
        if (k0 + BK < K) {
            int nxt = cur ^ 1;
#pragma unroll
            for (int i = 0; i < 4; ++i) {
                int row = c_row[i];
                ASYNC_COPY_B128(&Xq[(block_m + row) * (long long)K + (k0 + BK) + c_col],
                                &Asmem[nxt][row * BK + c_col]);
                ASYNC_COPY_B128(&Wq[(block_n + row) * (long long)K + (k0 + BK) + c_col],
                                &Bsmem[nxt][row * BK + c_col]);
            }
        }

        FragU afrag[2];
#pragma unroll
        for (int tm = 0; tm < 2; ++tm) {
            const unsigned char* abase = &Asmem[cur][(wm * 32 + tm * 16 + mrow) * BK];
#pragma unroll
            for (int g = 0; g < 4; ++g) {
                ((int4*)afrag[tm].d)[g] = *(const int4*)(abase + g * 32 + khi * 16);
            }
        }
        FragU bfrag[4];
#pragma unroll
        for (int tn = 0; tn < 4; ++tn) {
            const unsigned char* bbase = &Bsmem[cur][(wn * 64 + tn * 16 + mrow) * BK];
#pragma unroll
            for (int g = 0; g < 4; ++g) {
                ((int4*)bfrag[tn].d)[g] = *(const int4*)(bbase + g * 32 + khi * 16);
            }
        }

#pragma unroll
        for (int tm = 0; tm < 2; ++tm)
#pragma unroll
            for (int tn = 0; tn < 4; ++tn)
                acc[tm][tn] = __builtin_amdgcn_wmma_f32_16x16x128_fp8_fp8(
                    afrag[tm].v, bfrag[tn].v, (short)0, acc[tm][tn], false, false);

        WAIT_ASYNC();
        __syncthreads();
        cur ^= 1;
    }

    const float scale = (amax[0] * amax[1]) * (1.0f / (FP8_MAX * FP8_MAX));
#pragma unroll
    for (int tm = 0; tm < 2; ++tm) {
#pragma unroll
        for (int tn = 0; tn < 4; ++tn) {
            long long col  = block_n + wn * 64 + tn * 16 + mrow;
            long long row0 = block_m + wm * 32 + tm * 16 + khi * 8;
            float bv = bias[col];
#pragma unroll
            for (int r = 0; r < 8; ++r) {
                out[(row0 + r) * (long long)N + col] = acc[tm][tn][r] * scale + bv;
            }
        }
    }
}

extern "C" void kernel_launch(void* const* d_in, const int* in_sizes, int n_in,
                              void* d_out, int out_size, void* d_ws, size_t ws_size,
                              hipStream_t stream) {
    const float* x    = (const float*)d_in[0];
    const float* w    = (const float*)d_in[1];
    const float* bias = (const float*)d_in[2];
    float* out = (float*)d_out;

    const long long nx = in_sizes[0];
    const long long nw = in_sizes[1];
    const int N = in_sizes[2];
    const int K = (int)(nw / N);
    const int M = (int)(nx / K);

    float* amax = (float*)d_ws;
    unsigned char* xq = (unsigned char*)d_ws + 256;
    unsigned char* wq = xq + (size_t)M * (size_t)K;

    zero_scales_kernel<<<1, 32, 0, stream>>>(amax);
    amax_kernel<<<1024, 256, 0, stream>>>(x, nx / 4, amax + 0);
    amax_kernel<<<512,  256, 0, stream>>>(w, nw / 4, amax + 1);
    quant_kernel<<<2048, 256, 0, stream>>>(x, xq, nx / 4, amax + 0);
    quant_kernel<<<1024, 256, 0, stream>>>(w, wq, nw / 4, amax + 1);

    dim3 grid(N / BN, M / BM);
    gemm_fp8_kernel<<<grid, 256, 0, stream>>>(xq, wq, bias, amax, out, M, N, K);
}
